// MultiHeadGraphAttention_45861660787162
// MI455X (gfx1250) — matrix operand order repaired and checked by reading, not compiled, not run on hardware
//
#include <hip/hip_runtime.h>

typedef __attribute__((ext_vector_type(16))) _Float16 v16h;
typedef __attribute__((ext_vector_type(8)))  _Float16 v8h;
typedef __attribute__((ext_vector_type(8)))  float    v8f;

constexpr int kB = 4, kN = 2048, kD = 256, kH = 8, kDK = 32;
constexpr float kInvScale = 0.17677669529663687f; // 1/sqrt(32)

static __device__ __forceinline__ v8f wmma16x16x32(v16h a, v16h b, v8f c) {
  return __builtin_amdgcn_wmma_f32_16x16x32_f16(false, a, false, b, (short)0, c,
                                                false, false);
}

// ---------------------------------------------------------------------------
// Kernel 1: Q/K/V projection (f32 x, f32 W -> f16), one wave per 16x16 tile.
// Q,K stored [B,H,N,DK]; V stored transposed [B,H,DK,N].
// ---------------------------------------------------------------------------
__global__ __launch_bounds__(32) void qkv_proj_kernel(
    const float* __restrict__ x, const float* __restrict__ wq,
    const float* __restrict__ wk, const float* __restrict__ wv,
    _Float16* __restrict__ Qws, _Float16* __restrict__ Kws,
    _Float16* __restrict__ Vt) {
  const int lane = threadIdx.x & 31;
  const int half = lane >> 4;
  const int l    = lane & 15;
  const int n0 = blockIdx.x * 16;
  const int d0 = blockIdx.y * 16;
  const int b     = blockIdx.z / 3;
  const int which = blockIdx.z % 3;
  const float* W = (which == 0) ? wq : ((which == 1) ? wk : wv);

  const int rowA = n0 + l;
  const int colB = d0 + l;
  const float* xrow = x + ((size_t)b * kN + rowA) * kD;
  const float* wrow = W + (size_t)colB * kD;

  v8f acc = {};
  for (int kk = 0; kk < kD; kk += 32) {
    v16h a, bf;
    const float* ap = xrow + kk + half * 16;
#pragma unroll
    for (int j = 0; j < 8; ++j) {
      a[j]     = (_Float16)ap[j];
      a[j + 8] = (_Float16)ap[8 + j];
    }
    const float* bp = wrow + kk + half * 16;
#pragma unroll
    for (int j = 0; j < 16; ++j) bf[j] = (_Float16)bp[j];
    acc = wmma16x16x32(a, bf, acc);
  }

  // Branch-free store: select (uniform) destination base + stride once.
  const int h  = colB >> 5;
  const int dk = colB & 31;
  const size_t bh = (size_t)b * kH + h;
  _Float16* dstp;
  size_t base;
  size_t stride;
  if (which == 2) {  // V transposed: [B,H,DK,N], row index varies along n
    dstp   = Vt;
    base   = (bh * kDK + dk) * kN + (size_t)(n0 + half * 8);
    stride = 1;
  } else {           // Q/K: [B,H,N,DK], row index varies along n with stride DK
    dstp   = (which == 0) ? Qws : Kws;
    base   = (bh * kN + (size_t)(n0 + half * 8)) * kDK + dk;
    stride = kDK;
  }
#pragma unroll
  for (int r = 0; r < 8; ++r) dstp[base + (size_t)r * stride] = (_Float16)acc[r];
}

// ---------------------------------------------------------------------------
// Kernel 2: masked-softmax attention, single pass (online/flash softmax).
// One block (8 waves) per (b, h, 16 rows). Each wave owns 64-column m-chunks
// (amortizes the shfl row-max reduction + accumulator rescale over 4 score
// tiles), keeps running row max / sum, rescales its P*V accumulators on max
// updates. Cross-wave merge via the standard flash combine in LDS.
// ---------------------------------------------------------------------------
__global__ __launch_bounds__(256) void attn_kernel(
    const float* __restrict__ edge, const int* __restrict__ mask,
    const _Float16* __restrict__ Qws, const _Float16* __restrict__ Kws,
    const _Float16* __restrict__ Vt, _Float16* __restrict__ AO) {
  const int tid  = threadIdx.x;
  const int wid  = tid >> 5;
  const int lane = tid & 31;
  const int half = lane >> 4;
  const int l    = lane & 15;
  const int n0 = blockIdx.x * 16;
  const int h  = blockIdx.y;
  const int b  = blockIdx.z;

  __shared__ float    s_wmax[8][16];
  __shared__ float    s_wsum[8][16];
  __shared__ float    s_scale[8][16];
  __shared__ float    s_rowsum[16];
  __shared__ _Float16 s_stage[8][16][72];  // 72 f16 = 36 dwords: conflict-free
  __shared__ float    s_acc[8][16][32];

  const size_t bh = (size_t)b * kH + h;

  // Q fragment (A layout), reused for the whole row-tile.
  v16h qa;
  {
    const _Float16* qrow = Qws + (bh * kN + (n0 + l)) * kDK + half * 16;
    v8h lo = *(const v8h*)(qrow);
    v8h hi = *(const v8h*)(qrow + 8);
#pragma unroll
    for (int j = 0; j < 8; ++j) { qa[j] = lo[j]; qa[j + 8] = hi[j]; }
  }

  const float* erow_base = edge + ((size_t)b * kN + n0) * kN;
  const int*   mrow_base = mask + ((size_t)b * kN + n0) * kN;

  float run[8], rs[8];
#pragma unroll
  for (int r = 0; r < 8; ++r) { run[r] = -3.0e38f; rs[r] = 0.0f; }
  v8f acc0 = {}, acc1 = {};

  for (int t = wid; t < kN / 64; t += 8) {  // exactly 4 iterations per wave
    const int m0 = t * 64;

    // Light prefetch of next chunk's edge/mask stripe (global_prefetch_b8).
    if (t + 8 < kN / 64) {
      const size_t pofs = (size_t)(half * 8) * kN + (size_t)((t + 8) * 64 + l);
      __builtin_prefetch(erow_base + pofs, 0, 0);
      __builtin_prefetch(mrow_base + pofs, 0, 0);
    }

    // ---- scores for four 16-column tiles (C layout) ----
    float sc[4][8];
#pragma unroll
    for (int sub = 0; sub < 4; ++sub) {
      const int ms = m0 + sub * 16;
      v16h kf;
      {
        const _Float16* kr = Kws + (bh * kN + (ms + l)) * kDK + half * 16;
        v8h lo = *(const v8h*)(kr);
        v8h hi = *(const v8h*)(kr + 8);
#pragma unroll
        for (int j = 0; j < 8; ++j) { kf[j] = lo[j]; kf[j + 8] = hi[j]; }
      }
      v8f c = {};
      c = wmma16x16x32(qa, kf, c);
      const int m = ms + l;
#pragma unroll
      for (int r = 0; r < 8; ++r) {
        const int row = r + half * 8;
        const float e = erow_base[(size_t)row * kN + m];
        const int  mk = mrow_base[(size_t)row * kN + m];
        sc[sub][r] = (mk == 0) ? -1.0e9f : (c[r] * kInvScale + e);
      }
    }

    // ---- online softmax update (once per 64 columns) ----
    float tm[8];
#pragma unroll
    for (int r = 0; r < 8; ++r)
      tm[r] = fmaxf(fmaxf(sc[0][r], sc[1][r]), fmaxf(sc[2][r], sc[3][r]));
#pragma unroll
    for (int r = 0; r < 8; ++r) {
#pragma unroll
      for (int xm = 1; xm < 16; xm <<= 1)
        tm[r] = fmaxf(tm[r], __shfl_xor(tm[r], xm, 32));
    }
#pragma unroll
    for (int r = 0; r < 8; ++r) {
      const float nm = fmaxf(run[r], tm[r]);
      const float f  = __expf(run[r] - nm);  // 0 on first chunk, else <= 1
      run[r] = nm;
      rs[r]   *= f;
      acc0[r] *= f;
      acc1[r] *= f;
    }
#pragma unroll
    for (int sub = 0; sub < 4; ++sub) {
#pragma unroll
      for (int r = 0; r < 8; ++r) {
        const int row = r + half * 8;
        const float p = __expf(sc[sub][r] - run[r]);  // arg <= 0 always
        rs[r] += p;
        s_stage[wid][row][sub * 16 + l] = (_Float16)p;  // C-layout -> LDS
      }
    }

    // ---- P*V over two 32-wide K-chunks (same-wave LDS staging, in-order) ----
#pragma unroll
    for (int kc = 0; kc < 2; ++kc) {
      v16h av;
      {
        const _Float16* sp = &s_stage[wid][l][kc * 32 + half * 16];
        v8h lo = *(const v8h*)(sp);
        v8h hi = *(const v8h*)(sp + 8);
#pragma unroll
        for (int j = 0; j < 8; ++j) { av[j] = lo[j]; av[j + 8] = hi[j]; }
      }
      const int mc = m0 + kc * 32;
      const _Float16* vb0 = Vt + (bh * kDK + l)      * kN + mc + half * 16;
      const _Float16* vb1 = Vt + (bh * kDK + 16 + l) * kN + mc + half * 16;
      v8h a0 = *(const v8h*)(vb0);
      v8h a1 = *(const v8h*)(vb0 + 8);
      v8h b0 = *(const v8h*)(vb1);
      v8h b1 = *(const v8h*)(vb1 + 8);
      v16h v0, v1;
#pragma unroll
      for (int j = 0; j < 8; ++j) {
        v0[j] = a0[j]; v0[j + 8] = a1[j];
        v1[j] = b0[j]; v1[j + 8] = b1[j];
      }
      acc0 = wmma16x16x32(av, v0, acc0);
      acc1 = wmma16x16x32(av, v1, acc1);
    }
  }

  // ---- publish per-wave stats + partial accumulators ----
  if (l == 0) {
#pragma unroll
    for (int r = 0; r < 8; ++r) {
      s_wmax[wid][r + half * 8] = run[r];
      s_wsum[wid][r + half * 8] = rs[r];
    }
  }
#pragma unroll
  for (int r = 0; r < 8; ++r) {
    const int row = r + half * 8;
    s_acc[wid][row][l]      = acc0[r];
    s_acc[wid][row][16 + l] = acc1[r];
  }
  __syncthreads();

  // ---- flash combine across the 8 waves ----
  if (tid < 16) {
    float m = -3.0e38f;
#pragma unroll
    for (int w = 0; w < 8; ++w) m = fmaxf(m, s_wmax[w][tid]);
    float s = 0.0f;
#pragma unroll
    for (int w = 0; w < 8; ++w) {
      const float f = __expf(s_wmax[w][tid] - m);
      s_scale[w][tid] = f;
      s += s_wsum[w][tid] * f;
    }
    s_rowsum[tid] = s;
  }
  __syncthreads();

  for (int i = tid; i < 16 * 32; i += 256) {
    const int row = i >> 5;
    const int d   = i & 31;
    float s = 0.0f;
#pragma unroll
    for (int w = 0; w < 8; ++w) s += s_acc[w][row][d] * s_scale[w][row];
    s /= s_rowsum[row];
    AO[(((size_t)b * kN) + n0 + row) * kD + h * kDK + d] = (_Float16)s;
  }
}

// ---------------------------------------------------------------------------
// Kernel 3: output projection  out = AO @ w_o^T + b_o  (f32 result)
// ---------------------------------------------------------------------------
__global__ __launch_bounds__(32) void out_proj_kernel(
    const _Float16* __restrict__ AO, const float* __restrict__ wo,
    const float* __restrict__ bo, float* __restrict__ out) {
  const int lane = threadIdx.x & 31;
  const int half = lane >> 4;
  const int l    = lane & 15;
  const int n0 = blockIdx.x * 16;
  const int d0 = blockIdx.y * 16;
  const int b  = blockIdx.z;

  const int colB = d0 + l;
  const _Float16* arow = AO + ((size_t)b * kN + n0 + l) * kD;
  const float* wrow = wo + (size_t)colB * kD;

  v8f acc = {};
  for (int kk = 0; kk < kD; kk += 32) {
    v16h a, bf;
    const _Float16* ap = arow + kk + half * 16;
    v8h lo = *(const v8h*)ap;
    v8h hi = *(const v8h*)(ap + 8);
#pragma unroll
    for (int j = 0; j < 8; ++j) { a[j] = lo[j]; a[j + 8] = hi[j]; }
    const float* bp = wrow + kk + half * 16;
#pragma unroll
    for (int j = 0; j < 16; ++j) bf[j] = (_Float16)bp[j];
    acc = wmma16x16x32(a, bf, acc);
  }
  const float bias = bo[colB];
#pragma unroll
  for (int r = 0; r < 8; ++r) {
    const int n = n0 + r + half * 8;
    out[((size_t)b * kN + n) * kD + colB] = acc[r] + bias;
  }
}

// ---------------------------------------------------------------------------
extern "C" void kernel_launch(void* const* d_in, const int* in_sizes, int n_in,
                              void* d_out, int out_size, void* d_ws,
                              size_t ws_size, hipStream_t stream) {
  const float* x    = (const float*)d_in[0];
  const float* edge = (const float*)d_in[1];
  const int*   mask = (const int*)d_in[2];
  const float* wq   = (const float*)d_in[3];
  const float* wk   = (const float*)d_in[4];
  const float* wv   = (const float*)d_in[5];
  const float* wo   = (const float*)d_in[6];
  const float* bo   = (const float*)d_in[7];
  float* out = (float*)d_out;

  _Float16* Qws = (_Float16*)d_ws;                       // [B,H,N,DK]
  _Float16* Kws = Qws + (size_t)kB * kH * kN * kDK;      // [B,H,N,DK]
  _Float16* Vt  = Kws + (size_t)kB * kH * kN * kDK;      // [B,H,DK,N]
  _Float16* AO  = Vt  + (size_t)kB * kH * kN * kDK;      // [B,N,D]

  dim3 g1(kN / 16, kD / 16, kB * 3);
  qkv_proj_kernel<<<g1, 32, 0, stream>>>(x, wq, wk, wv, Qws, Kws, Vt);

  dim3 g2(kN / 16, kH, kB);
  attn_kernel<<<g2, 256, 0, stream>>>(edge, mask, Qws, Kws, Vt, AO);

  dim3 g3(kN / 16, kD / 16, kB);
  out_proj_kernel<<<g3, 32, 0, stream>>>(AO, wo, bo, out);
}
